// WindowAttention_71356586655755
// MI455X (gfx1250) — matrix operand order repaired and checked by reading, not compiled, not run on hardware
//
#include <hip/hip_runtime.h>
#include <hip/hip_bf16.h>

typedef __attribute__((ext_vector_type(16))) _Float16 v16h;
typedef __attribute__((ext_vector_type(8)))  float    v8f;
typedef __attribute__((ext_vector_type(4)))  _Float16 v4h;

namespace {
constexpr int   kB    = 8192;
constexpr int   kN    = 64;    // tokens per window
constexpr int   kC    = 128;   // DIM
constexpr int   kH    = 4;     // heads
constexpr int   kD    = 32;    // head dim
constexpr int   kTab  = 225;   // (2*8-1)^2
constexpr float kScale = 0.17677669529663687f;  // 32^-0.5

// workspace layout (u32 units)
constexpr int kQkvFragDw  = 4 * 24 * 256;  // ktiles*ntiles*(32 lanes*8 dw) = 24576
constexpr int kProjFragDw = 4 *  8 * 256;  // 8192
constexpr int kBiasF      = kH * kN * kN;  // 16384
}

__device__ __forceinline__ v8f wmma_f32_f16(v16h a, v16h b, v8f c) {
  return __builtin_amdgcn_wmma_f32_16x16x32_f16(false, a, false, b, (short)0, c,
                                                false, false);
}

// A fragment: 16x32 f16 tile from a row-major matrix (leading dim ld, halfs).
// Contiguous k order: lane(half,m): vI holds K = half*16 + 2i,2i+1.
__device__ __forceinline__ v16h load_a_frag(const _Float16* base, int ld) {
  const int lane = threadIdx.x & 31;
  const int half = lane >> 4;
  const int m    = lane & 15;
  union { v16h h; unsigned u[8]; } r;
  const _Float16* row = base + m * ld;
#pragma unroll
  for (int i = 0; i < 4; ++i)
    r.u[i] = *(const unsigned*)(row + half * 16 + 2 * i);
#pragma unroll
  for (int i = 0; i < 4; ++i)
    r.u[4 + i] = *(const unsigned*)(row + 8 + half * 16 + 2 * i);
  return r.h;
}

// B fragment (32x16, B[k][n]) built from a row-major matrix holding B^T,
// i.e. mat[n][k] = B[k][n]; base points at (n=0, k=0) of the tile.
// Contiguous k order: vI lane(half,n): K = half*16 + 2i, 2i+1.
__device__ __forceinline__ v16h load_bT_frag(const _Float16* base, int ld) {
  const int lane = threadIdx.x & 31;
  const int half = lane >> 4;
  const int n    = lane & 15;
  union { v16h h; unsigned u[8]; } r;
  const _Float16* row = base + n * ld + half * 16;
#pragma unroll
  for (int i = 0; i < 8; ++i)
    r.u[i] = *(const unsigned*)(row + 2 * i);
  return r.h;
}

// ---------------- prep: weights -> f16 WMMA-B-fragment layout ---------------
__global__ void prep_frag_kernel(const float* __restrict__ W,
                                 unsigned* __restrict__ out,
                                 int ldn, int ntiles, int total) {
  int idx = blockIdx.x * blockDim.x + threadIdx.x;
  if (idx >= total) return;
  int tile = idx >> 8;          // (kt*ntiles + nt)
  int rem  = idx & 255;
  int lane = rem >> 3;
  int i    = rem & 7;
  int half = lane >> 4;
  int n    = lane & 15;
  int kt   = tile / ntiles;
  int nt   = tile - kt * ntiles;
  int k0   = kt * 32 + half * 16 + 2 * i;
  int col  = nt * 16 + n;
  union { _Float16 h[2]; unsigned u; } p;
  p.h[0] = (_Float16)W[(size_t)k0 * ldn + col];
  p.h[1] = (_Float16)W[(size_t)(k0 + 1) * ldn + col];
  out[idx] = p.u;
}

// ---------------- prep: gather relative-position bias -----------------------
__global__ void prep_bias_kernel(const float* __restrict__ table,
                                 const int* __restrict__ rel,
                                 float* __restrict__ out) {
  int idx = blockIdx.x * blockDim.x + threadIdx.x;
  if (idx >= kBiasF) return;
  int h  = idx >> 12;           // / 4096
  int nm = idx & 4095;
  out[idx] = table[h * kTab + rel[nm]];
}

// ---------------- main: one workgroup (8 waves) per window ------------------
__global__ __launch_bounds__(256)
void win_attn_kernel(const float* __restrict__ x,
                     const float* __restrict__ mask,
                     const float* __restrict__ qkv_b,
                     const float* __restrict__ proj_b,
                     const unsigned* __restrict__ qkv_frag,
                     const unsigned* __restrict__ proj_frag,
                     const float* __restrict__ bias_hnm,
                     float* __restrict__ out) {
  __shared__ __align__(16) _Float16 lds_x [kN * kC];        // X, later O (16KB)
  __shared__ __align__(16) _Float16 lds_qk[kN * 256];       // Q|K, later P (32KB)
  __shared__ __align__(16) _Float16 lds_vt[kH * kD * kN];   // V^T per head (16KB)
  __shared__ __align__(16) float    lds_s [kH * kN * kN];   // scores f32 (64KB)

  const int b    = blockIdx.x;
  const int tid  = threadIdx.x;
  const int lane = tid & 31;
  const int wave = tid >> 5;          // 0..7
  const int half = lane >> 4;
  const int nn   = lane & 15;

  // ---- phase 1: stage X into LDS as f16 ; prefetch weight fragments ----
  {
    const float4* xb = (const float4*)(x + (size_t)b * kN * kC);
    v4h* dst = (v4h*)lds_x;
    for (int i = tid; i < kN * kC / 4; i += 256) {
      float4 v = xb[i];
      v4h h;
      h[0] = (_Float16)v.x; h[1] = (_Float16)v.y;
      h[2] = (_Float16)v.z; h[3] = (_Float16)v.w;
      dst[i] = h;
    }
    // warm L0/L2 with the (L2-resident) pre-swizzled weights
    for (int i = tid * 64; i < kQkvFragDw; i += 256 * 64)
      __builtin_prefetch(qkv_frag + i, 0, 3);
    for (int i = tid * 64; i < kProjFragDw; i += 256 * 64)
      __builtin_prefetch(proj_frag + i, 0, 3);
  }
  __syncthreads();

  // ---- phase 2: QKV = X @ Wqkv + b ----
  // wave w: mt = w>>1 (fixed -> A frags loaded once), nt = (w&1) + 2*j.
  {
    const v16h* wf = (const v16h*)qkv_frag;
    const int mt = wave >> 1;
    v16h a0 = load_a_frag(lds_x + mt * 16 * kC +  0, kC);
    v16h a1 = load_a_frag(lds_x + mt * 16 * kC + 32, kC);
    v16h a2 = load_a_frag(lds_x + mt * 16 * kC + 64, kC);
    v16h a3 = load_a_frag(lds_x + mt * 16 * kC + 96, kC);
    for (int nt = (wave & 1); nt < 24; nt += 2) {
      v8f c = {};
      c = wmma_f32_f16(a0, wf[(0 * 24 + nt) * 32 + lane], c);
      c = wmma_f32_f16(a1, wf[(1 * 24 + nt) * 32 + lane], c);
      c = wmma_f32_f16(a2, wf[(2 * 24 + nt) * 32 + lane], c);
      c = wmma_f32_f16(a3, wf[(3 * 24 + nt) * 32 + lane], c);
      const int col  = nt * 16 + nn;
      const float bv = qkv_b[col];
      if (nt < 16) {   // Q (cols 0..127) | K (cols 128..255), row-major
#pragma unroll
        for (int r = 0; r < 8; ++r)
          lds_qk[(mt * 16 + half * 8 + r) * 256 + col] = (_Float16)(c[r] + bv);
      } else {         // V -> store transposed per head: vt[h][d][token]
        const int vcol = col - 256;
        const int h = vcol >> 5, d = vcol & 31;
#pragma unroll
        for (int r = 0; r < 8; ++r)
          lds_vt[(h * kD + d) * kN + mt * 16 + half * 8 + r] =
              (_Float16)(c[r] + bv);
      }
    }
  }
  __syncthreads();

  // ---- phase 3: S = scale*(Q K^T) + bias + mask (fused, coalesced adds) ----
  // wave w: head h = w>>1, mt in {(w&1)*2, (w&1)*2+1}, nt = 0..3.
  {
    const int h = wave >> 1;
    const float* mb = mask + (size_t)b * kN * kN;
    const float* bb = bias_hnm + h * kN * kN;
    for (int mi = 0; mi < 2; ++mi) {
      const int mt = (wave & 1) * 2 + mi;
      v16h a = load_a_frag(lds_qk + (mt * 16) * 256 + h * kD, 256);
      for (int nt = 0; nt < 4; ++nt) {
        v16h bt = load_bT_frag(lds_qk + (nt * 16) * 256 + 128 + h * kD, 256);
        v8f c = {};
        c = wmma_f32_f16(a, bt, c);
#pragma unroll
        for (int r = 0; r < 8; ++r) {
          const int row = mt * 16 + half * 8 + r;
          const int col = nt * 16 + nn;
          lds_s[(h * kN + row) * kN + col] =
              c[r] * kScale + bb[row * kN + col] + mb[row * kN + col];
        }
      }
    }
  }
  __syncthreads();

  // ---- phase 4: softmax over LDS only; P as f16 over dead Q|K ----
  _Float16* lds_p = lds_qk;
  {
    const int h   = tid >> 6;       // 256 threads -> one row per thread
    const int row = tid & 63;
    float* srow = lds_s + (h * kN + row) * kN;
    float mx = -3.4e38f;
    for (int m = 0; m < kN; ++m) mx = fmaxf(mx, srow[m]);
    _Float16* prow = lds_p + (h * kN + row) * kN;
    float sum = 0.f;
    for (int m = 0; m < kN; ++m) {
      float e = __expf(srow[m] - mx);
      sum += e;
      prow[m] = (_Float16)e;
    }
    const float inv = 1.f / sum;
    for (int m = 0; m < kN; ++m)
      prow[m] = (_Float16)((float)prow[m] * inv);
  }
  __syncthreads();

  // ---- phase 5: O = P V (O overwrites dead X) ----
  // wave w: head h = w>>1, mt in {(w&1)*2, (w&1)*2+1}, nt = 0..1.
  {
    const int h = wave >> 1;
    for (int mi = 0; mi < 2; ++mi) {
      const int mt = (wave & 1) * 2 + mi;
      v16h a0 = load_a_frag(lds_p + (h * kN + mt * 16) * kN +  0, kN);
      v16h a1 = load_a_frag(lds_p + (h * kN + mt * 16) * kN + 32, kN);
      for (int nt = 0; nt < 2; ++nt) {
        v16h b0 = load_bT_frag(lds_vt + (h * kD + nt * 16) * kN +  0, kN);
        v16h b1 = load_bT_frag(lds_vt + (h * kD + nt * 16) * kN + 32, kN);
        v8f c = {};
        c = wmma_f32_f16(a0, b0, c);
        c = wmma_f32_f16(a1, b1, c);
        const int col = h * kD + nt * 16 + nn;
#pragma unroll
        for (int r = 0; r < 8; ++r)
          lds_x[(mt * 16 + half * 8 + r) * kC + col] = (_Float16)c[r];
      }
    }
  }
  __syncthreads();

  // ---- phase 6: out = O @ Wproj + b  (f32 to HBM) ----
  // wave w: mt = w>>1 (A frags loaded once), nt = (w&1)*4 .. +3.
  {
    const v16h* wf = (const v16h*)proj_frag;
    float* ob = out + (size_t)b * kN * kC;
    const int mt = wave >> 1;
    v16h a0 = load_a_frag(lds_x + mt * 16 * kC +  0, kC);
    v16h a1 = load_a_frag(lds_x + mt * 16 * kC + 32, kC);
    v16h a2 = load_a_frag(lds_x + mt * 16 * kC + 64, kC);
    v16h a3 = load_a_frag(lds_x + mt * 16 * kC + 96, kC);
    for (int j = 0; j < 4; ++j) {
      const int nt = (wave & 1) * 4 + j;
      v8f c = {};
      c = wmma_f32_f16(a0, wf[(0 * 8 + nt) * 32 + lane], c);
      c = wmma_f32_f16(a1, wf[(1 * 8 + nt) * 32 + lane], c);
      c = wmma_f32_f16(a2, wf[(2 * 8 + nt) * 32 + lane], c);
      c = wmma_f32_f16(a3, wf[(3 * 8 + nt) * 32 + lane], c);
      const int col  = nt * 16 + nn;
      const float pb = proj_b[col];
#pragma unroll
      for (int r = 0; r < 8; ++r)
        ob[(mt * 16 + half * 8 + r) * kC + col] = c[r] + pb;
    }
  }
}

extern "C" void kernel_launch(void* const* d_in, const int* in_sizes, int n_in,
                              void* d_out, int out_size, void* d_ws, size_t ws_size,
                              hipStream_t stream) {
  const float* x      = (const float*)d_in[0];
  const float* mask   = (const float*)d_in[1];
  const float* qkv_w  = (const float*)d_in[2];
  const float* qkv_b  = (const float*)d_in[3];
  const float* proj_w = (const float*)d_in[4];
  const float* proj_b = (const float*)d_in[5];
  const float* btab   = (const float*)d_in[6];
  const int*   relidx = (const int*)d_in[7];
  float* out = (float*)d_out;

  unsigned* qkvf  = (unsigned*)d_ws;
  unsigned* projf = qkvf + kQkvFragDw;
  float*    biasf = (float*)(projf + kProjFragDw);

  prep_frag_kernel<<<(kQkvFragDw + 255) / 256, 256, 0, stream>>>(
      qkv_w, qkvf, 3 * kC, 24, kQkvFragDw);
  prep_frag_kernel<<<(kProjFragDw + 255) / 256, 256, 0, stream>>>(
      proj_w, projf, kC, 8, kProjFragDw);
  prep_bias_kernel<<<(kBiasF + 255) / 256, 256, 0, stream>>>(btab, relidx, biasf);

  win_attn_kernel<<<kB, 256, 0, stream>>>(x, mask, qkv_b, proj_b,
                                          qkvf, projf, biasf, out);
}
